// FlashCrossAttention_20280835572118
// MI455X (gfx1250) — matrix operand order repaired and checked by reading, not compiled, not run on hardware
//
#include <hip/hip_runtime.h>

// ---------------------------------------------------------------------------
// FlashCrossAttention for MI455X (gfx1250): bf16 WMMA for all GEMM-shaped ops,
// fp32 accumulate, fp32 softmax/rmsnorm/rope. Async global->LDS tile copies.
// Shapes: B=4, Td=Te=2048, d=1024, H=16, D=64  -> M=B*T=8192.
// ---------------------------------------------------------------------------

typedef __attribute__((ext_vector_type(16))) __bf16        v16bf;
typedef __attribute__((ext_vector_type(8)))  float         v8f;
typedef __attribute__((ext_vector_type(8)))  unsigned int  v8u;

// pointee type for the async-LDS builtin: int __vector(4) (from hipcc diag)
typedef __attribute__((__vector_size__(16))) int i32x4;
typedef __attribute__((address_space(1))) i32x4 i32x4_as1;   // global
typedef __attribute__((address_space(3))) i32x4 i32x4_as3;   // LDS

#define WG_B   4
#define WG_T   2048
#define WG_D   1024
#define WG_H   16
#define WG_HD  64
#define WG_M   (WG_B * WG_T)

#if defined(__has_builtin)
#if __has_builtin(__builtin_amdgcn_global_load_async_to_lds_b128)
#define HAVE_ASYNC_LDS 1
#endif
#endif

// ---------------- async global->LDS 16B copy -------------------------------
__device__ __forceinline__ void async_copy16(const unsigned short* g,
                                             unsigned short* l) {
#ifdef HAVE_ASYNC_LDS
  __builtin_amdgcn_global_load_async_to_lds_b128(
      (i32x4_as1*)g, (i32x4_as3*)l, /*imm offset=*/0, /*cpol=*/0);
#else
  *(uint4*)l = *(const uint4*)g;
#endif
}
__device__ __forceinline__ void wait_async_lds() {
#ifdef HAVE_ASYNC_LDS
#if __has_builtin(__builtin_amdgcn_s_wait_asynccnt)
  __builtin_amdgcn_s_wait_asynccnt(0);
#else
  asm volatile("s_wait_asynccnt 0x0" ::: "memory");
#endif
#endif
}

// ---------------- bf16 helpers (raw u16 storage, RNE convert) --------------
__device__ __forceinline__ unsigned short f32_to_bf16(float f) {
  union { float f; unsigned int u; } v; v.f = f;
  unsigned int r = v.u + 0x7FFFu + ((v.u >> 16) & 1u);
  return (unsigned short)(r >> 16);
}
__device__ __forceinline__ float bf16_to_f32(unsigned short h) {
  union { unsigned int u; float f; } v; v.u = ((unsigned int)h) << 16;
  return v.f;
}

// ---------------- WMMA fragment loaders ------------------------------------
// B-fragment (32x16 KxN bf16): lane n=lane&15, half=lane>>4; 16 contiguous K
// values starting at half*16 (this code's k order, not ISA table order) -> 2x b128.
__device__ __forceinline__ v16bf frag_contig(const unsigned short* p) {
  const uint4* q = (const uint4*)p;
  uint4 a = q[0], b = q[1];
  v8u u; u[0]=a.x; u[1]=a.y; u[2]=a.z; u[3]=a.w; u[4]=b.x; u[5]=b.y; u[6]=b.z; u[7]=b.w;
  return __builtin_bit_cast(v16bf, u);
}
// A-fragment (16x32 MxK bf16): lane m=lane&15, half=lane>>4; K covers
// {half*16..+15} contiguous (matches B fragment k order) -> 2x b128.
__device__ __forceinline__ v16bf frag_split(const unsigned short* p, int hl) {
  const uint4* q0 = (const uint4*)(p + hl * 16);
  const uint4* q1 = (const uint4*)(p + hl * 16 + 8);
  uint4 a = *q0, b = *q1;
  v8u u; u[0]=a.x; u[1]=a.y; u[2]=a.z; u[3]=a.w; u[4]=b.x; u[5]=b.y; u[6]=b.z; u[7]=b.w;
  return __builtin_bit_cast(v16bf, u);
}
__device__ __forceinline__ v8f wmma_bf16(v16bf a, v16bf b, v8f c) {
  return __builtin_amdgcn_wmma_f32_16x16x32_bf16(
      /*neg_a=*/false, a, /*neg_b=*/false, b,
      /*c_mod=*/(short)0, c, /*reuse_a=*/false, /*reuse_b=*/false);
}

// ---------------- RMSNorm -> bf16 ------------------------------------------
__global__ void rmsnorm_to_bf16_k(const float* __restrict__ x,
                                  const float* __restrict__ w,
                                  unsigned short* __restrict__ out, int d) {
  const int row = blockIdx.x;
  const int tid = threadIdx.x;
  const float* xr = x + (size_t)row * d;
  __shared__ float red[256];
  float ss = 0.f;
  for (int i = tid; i < d; i += 256) { float v = xr[i]; ss += v * v; }
  red[tid] = ss; __syncthreads();
  for (int s = 128; s > 0; s >>= 1) {
    if (tid < s) red[tid] += red[tid + s];
    __syncthreads();
  }
  const float inv = rsqrtf(red[0] / (float)d + 1e-6f);
  for (int i = tid; i < d; i += 256)
    out[(size_t)row * d + i] = f32_to_bf16(xr[i] * inv * w[i]);
}

// ---------------- f32 -> bf16 convert --------------------------------------
__global__ void conv_bf16_k(const float* __restrict__ in,
                            unsigned short* __restrict__ out, int n) {
  int i = blockIdx.x * blockDim.x + threadIdx.x;
  if (i < n) out[i] = f32_to_bf16(in[i]);
}

// ---------------- RoPE + [B,T,H,D] -> [B,H,T,D] transpose ------------------
__global__ void rope_transpose_k(const unsigned short* __restrict__ src,
                                 int srcStride, int srcColOff,
                                 unsigned short* __restrict__ dst,
                                 int T, int doRope,
                                 const int* __restrict__ offPtr, int useOff) {
  const long idx = (long)blockIdx.x * blockDim.x + threadIdx.x;
  const int  i   = (int)(idx & 31);
  const long r   = idx >> 5;            // (b*H + h)*T + t
  const int  t   = (int)(r % T);
  const long bh  = r / T;
  const int  h   = (int)(bh & (WG_H - 1));
  const int  b   = (int)(bh >> 4);
  const long tok = (long)b * T + t;
  const unsigned short* s = src + tok * srcStride + srcColOff + h * WG_HD;
  float x1 = bf16_to_f32(s[i]);
  float x2 = bf16_to_f32(s[i + 32]);
  float o1, o2;
  if (doRope) {
    const int off = useOff ? *offPtr : 0;
    const float invf = __expf(-(float)i * (9.210340371976184f / 32.0f));
    const float ang  = (float)(t + off) * invf;
    float sn, cs; __sincosf(ang, &sn, &cs);
    o1 = x1 * cs - x2 * sn;
    o2 = x2 * cs + x1 * sn;
  } else { o1 = x1; o2 = x2; }
  unsigned short* d = dst + (bh * T + t) * WG_HD;
  d[i]      = f32_to_bf16(o1);
  d[i + 32] = f32_to_bf16(o2);
}

// ---------------- WMMA GEMM: C[M,N] = A[M,K] * W[N,K]^T --------------------
// Block 128 threads (4 waves), tile 64(M)x128(N), K-step 32.
// Wave: 32x64 => 2x4 WMMA tiles per k-step (8 wmma : 12 ds_load_b128).
// Tiles staged with global_load_async_to_lds_b128 (ASYNCcnt), addresses hoisted.
__global__ __launch_bounds__(128)
void gemm_bf16_wmma_k(const unsigned short* __restrict__ A,
                      const unsigned short* __restrict__ Bw,
                      int M, int N, int K,
                      unsigned short* __restrict__ out16,
                      float* __restrict__ outf,
                      const float* __restrict__ residual) {
  __shared__ unsigned short Al[64 * 32];    // [row][k]
  __shared__ unsigned short Bl[128 * 32];   // [row][k]
  const int tid  = threadIdx.x;
  const int wave = tid >> 5, lane = tid & 31, l15 = lane & 15, hl = lane >> 4;
  const int bm = blockIdx.y * 64, bn = blockIdx.x * 128;
  const int wm = (wave >> 1) * 32, wn = (wave & 1) * 64;

  // hoisted per-thread copy addresses: chunk c -> row=c>>2, seg=c&3
  const int crow = tid >> 2, cseg = (tid & 3) * 8;
  const unsigned short* gA0 = A  + (size_t)(bm + crow)      * K + cseg;
  const unsigned short* gA1 = A  + (size_t)(bm + crow + 32) * K + cseg;
  const unsigned short* gB0 = Bw + (size_t)(bn + crow)      * K + cseg;
  const unsigned short* gB1 = Bw + (size_t)(bn + crow + 32) * K + cseg;
  const unsigned short* gB2 = Bw + (size_t)(bn + crow + 64) * K + cseg;
  const unsigned short* gB3 = Bw + (size_t)(bn + crow + 96) * K + cseg;
  unsigned short* lA0 = Al + tid * 8;
  unsigned short* lA1 = Al + (tid + 128) * 8;
  unsigned short* lB0 = Bl + tid * 8;
  unsigned short* lB1 = Bl + (tid + 128) * 8;
  unsigned short* lB2 = Bl + (tid + 256) * 8;
  unsigned short* lB3 = Bl + (tid + 384) * 8;

  v8f acc[2][4] = {};

  for (int k0 = 0; k0 < K; k0 += 32) {
    __syncthreads();                         // prior iter frag reads complete
    async_copy16(gA0 + k0, lA0);
    async_copy16(gA1 + k0, lA1);
    async_copy16(gB0 + k0, lB0);
    async_copy16(gB1 + k0, lB1);
    async_copy16(gB2 + k0, lB2);
    async_copy16(gB3 + k0, lB3);
    wait_async_lds();
    __syncthreads();

    v16bf af[2], bfr[4];
    #pragma unroll
    for (int mt = 0; mt < 2; ++mt)
      af[mt] = frag_split(Al + (wm + mt * 16 + l15) * 32, hl);
    #pragma unroll
    for (int nt = 0; nt < 4; ++nt)
      bfr[nt] = frag_contig(Bl + (wn + nt * 16 + l15) * 32 + hl * 16);
    #pragma unroll
    for (int mt = 0; mt < 2; ++mt)
      #pragma unroll
      for (int nt = 0; nt < 4; ++nt)
        acc[mt][nt] = wmma_bf16(af[mt], bfr[nt], acc[mt][nt]);
  }

  #pragma unroll
  for (int mt = 0; mt < 2; ++mt)
    #pragma unroll
    for (int nt = 0; nt < 4; ++nt)
      #pragma unroll
      for (int r = 0; r < 8; ++r) {
        const int m = bm + wm + mt * 16 + r + 8 * hl;
        const int n = bn + wn + nt * 16 + l15;
        const float v = acc[mt][nt][r];
        if (outf) outf[(size_t)m * N + n] = residual[(size_t)m * N + n] + v;
        else      out16[(size_t)m * N + n] = f32_to_bf16(v);
      }
}

// ---------------- Flash attention core -------------------------------------
// Grid: (Td/64, B*H). Block 128 = 4 waves; wave handles 16 q rows.
// K tile staged via async global->LDS; V transposed on LDS store.
__global__ __launch_bounds__(128)
void flash_attn_k(const unsigned short* __restrict__ Q,
                  const unsigned short* __restrict__ Kr,
                  const unsigned short* __restrict__ Vr,
                  unsigned short* __restrict__ Oa,
                  int Td, int Te) {
  __shared__ unsigned short Kl[32 * 64];       // [key][dim]
  __shared__ unsigned short Vl[64 * 32];       // transposed: [dim][key]
  __shared__ unsigned short Pl[4 * 16 * 32];   // per-wave P staging
  const int tid  = threadIdx.x;
  const int wave = tid >> 5, lane = tid & 31, l15 = lane & 15, hl = lane >> 4;
  const int q0 = blockIdx.x * 64;
  const int bh = blockIdx.y;
  const int b  = bh >> 4, h = bh & (WG_H - 1);
  const unsigned short* Qb = Q  + (size_t)bh * Td * WG_HD;
  const unsigned short* Kb = Kr + (size_t)bh * Te * WG_HD;
  const unsigned short* Vb = Vr + (size_t)bh * Te * WG_HD;

  // Q fragments (16 rows x 64 dims = 2 K-chunks), loaded once from global.
  const unsigned short* qrow = Qb + (size_t)(q0 + wave * 16 + l15) * WG_HD;
  v16bf qf0 = frag_split(qrow,      hl);
  v16bf qf1 = frag_split(qrow + 32, hl);

  // hoisted copy addresses: chunk c -> key=c>>3, dseg=c&7
  const int krow = tid >> 3, kseg = (tid & 7) * 8;
  const unsigned short* gK = Kb + (size_t)krow * WG_HD + kseg;
  const unsigned short* gV = Vb + (size_t)krow * WG_HD + kseg;
  unsigned short* lK0 = Kl + tid * 8;
  unsigned short* lK1 = Kl + (tid + 128) * 8;

  v8f acc[4] = {};
  float runmax[8], runsum[8], alph[8];
  #pragma unroll
  for (int r = 0; r < 8; ++r) { runmax[r] = -3.0e38f; runsum[r] = 0.f; }
  const float scale = 0.125f;  // 1/sqrt(64)

  for (int t0 = 0; t0 < Te; t0 += 32) {
    __syncthreads();                      // prior iter Kl/Vl reads complete
    // K tile: async copy, row-major [key][dim]
    async_copy16(gK + (size_t)t0 * WG_HD, lK0);
    async_copy16(gK + (size_t)(t0 + 16) * WG_HD, lK1);
    // V tile: load + transpose on LDS store -> Vl[dim][key]
    uint4 v4a = *(const uint4*)(gV + (size_t)t0 * WG_HD);
    uint4 v4b = *(const uint4*)(gV + (size_t)(t0 + 16) * WG_HD);
    #pragma unroll
    for (int pass = 0; pass < 2; ++pass) {
      const uint4 v4 = pass ? v4b : v4a;
      const int key = krow + pass * 16;
      Vl[(kseg + 0) * 32 + key] = (unsigned short)(v4.x);
      Vl[(kseg + 1) * 32 + key] = (unsigned short)(v4.x >> 16);
      Vl[(kseg + 2) * 32 + key] = (unsigned short)(v4.y);
      Vl[(kseg + 3) * 32 + key] = (unsigned short)(v4.y >> 16);
      Vl[(kseg + 4) * 32 + key] = (unsigned short)(v4.z);
      Vl[(kseg + 5) * 32 + key] = (unsigned short)(v4.z >> 16);
      Vl[(kseg + 6) * 32 + key] = (unsigned short)(v4.w);
      Vl[(kseg + 7) * 32 + key] = (unsigned short)(v4.w >> 16);
    }
    // prefetch next tile while this one is consumed
    if (t0 + 32 < Te) {
      __builtin_prefetch(gK + (size_t)(t0 + 32) * WG_HD, 0, 1);
      __builtin_prefetch(gV + (size_t)(t0 + 32) * WG_HD, 0, 1);
    }
    wait_async_lds();
    __syncthreads();

    // S = Q @ K^T  (16 x 32 scores, two 16-key n-tiles)
    v8f s0 = {}, s1 = {};
    {
      v16bf b00 = frag_contig(Kl + (l15)      * 64 +  0 + hl * 16);
      v16bf b01 = frag_contig(Kl + (16 + l15) * 64 +  0 + hl * 16);
      s0 = wmma_bf16(qf0, b00, s0);
      s1 = wmma_bf16(qf0, b01, s1);
      v16bf b10 = frag_contig(Kl + (l15)      * 64 + 32 + hl * 16);
      v16bf b11 = frag_contig(Kl + (16 + l15) * 64 + 32 + hl * 16);
      s0 = wmma_bf16(qf1, b10, s0);
      s1 = wmma_bf16(qf1, b11, s1);
    }

    // online softmax (fp32): rows m = r + 8*hl live across a 16-lane group
    unsigned short* Pw = Pl + wave * 512;
    #pragma unroll
    for (int r = 0; r < 8; ++r) {
      float v0 = s0[r] * scale, v1 = s1[r] * scale;
      float mx = fmaxf(v0, v1);
      mx = fmaxf(mx, __shfl_xor(mx, 1));
      mx = fmaxf(mx, __shfl_xor(mx, 2));
      mx = fmaxf(mx, __shfl_xor(mx, 4));
      mx = fmaxf(mx, __shfl_xor(mx, 8));
      const float nm = fmaxf(runmax[r], mx);
      const float a  = __expf(runmax[r] - nm);
      const float p0 = __expf(v0 - nm);
      const float p1 = __expf(v1 - nm);
      float sm = p0 + p1;
      sm += __shfl_xor(sm, 1);
      sm += __shfl_xor(sm, 2);
      sm += __shfl_xor(sm, 4);
      sm += __shfl_xor(sm, 8);
      runsum[r] = runsum[r] * a + sm;
      runmax[r] = nm;
      alph[r]   = a;
      Pw[(r + 8 * hl) * 32 +      l15] = f32_to_bf16(p0);
      Pw[(r + 8 * hl) * 32 + 16 + l15] = f32_to_bf16(p1);
    }
    #pragma unroll
    for (int nt = 0; nt < 4; ++nt)
      #pragma unroll
      for (int r = 0; r < 8; ++r) acc[nt][r] *= alph[r];

    __syncthreads();                      // P visible (uniform barrier)

    // O += P @ V  (P: 16x32 A-frag from LDS bounce; V: 4 dim-tiles)
    v16bf pf = frag_split(Pw + l15 * 32, hl);
    #pragma unroll
    for (int nt = 0; nt < 4; ++nt) {
      v16bf vb = frag_contig(Vl + (nt * 16 + l15) * 32 + hl * 16);
      acc[nt] = wmma_bf16(pf, vb, acc[nt]);
    }
  }

  // normalize and write [B*T, H*64] bf16
  #pragma unroll
  for (int nt = 0; nt < 4; ++nt)
    #pragma unroll
    for (int r = 0; r < 8; ++r) {
      const int t   = q0 + wave * 16 + r + 8 * hl;
      const int col = h * WG_HD + nt * 16 + l15;
      Oa[(size_t)(b * Td + t) * WG_D + col] =
          f32_to_bf16(acc[nt][r] * (1.0f / runsum[r]));
    }
}

// ---------------------------------------------------------------------------
extern "C" void kernel_launch(void* const* d_in, const int* in_sizes, int n_in,
                              void* d_out, int out_size, void* d_ws, size_t ws_size,
                              hipStream_t stream) {
  const float* x    = (const float*)d_in[0];   // [4,2048,1024]
  const float* enc  = (const float*)d_in[1];   // [4,2048,1024]
  const float* nw   = (const float*)d_in[2];   // [1024]
  const float* Wq   = (const float*)d_in[3];   // [1024,1024]
  const float* Wkv  = (const float*)d_in[4];   // [2048,1024]
  const float* Wo   = (const float*)d_in[5];   // [1024,1024]
  const int*   doff = (const int*)d_in[6];     // scalar
  float* out = (float*)d_out;

  char* ws = (char*)d_ws;
  const size_t MB = 1024ull * 1024ull;
  unsigned short* h16    = (unsigned short*)(ws +  0 * MB);  // 16MB (later qr16)
  unsigned short* enc16  = (unsigned short*)(ws + 16 * MB);  // 16MB (later kr16)
  unsigned short* wq16   = (unsigned short*)(ws + 32 * MB);  //  2MB
  unsigned short* wkv16  = (unsigned short*)(ws + 34 * MB);  //  4MB
  unsigned short* wo16   = (unsigned short*)(ws + 38 * MB);  //  2MB
  unsigned short* qraw   = (unsigned short*)(ws + 40 * MB);  // 16MB (later attn16)
  unsigned short* kvraw  = (unsigned short*)(ws + 56 * MB);  // 32MB
  unsigned short* v16p   = (unsigned short*)(ws + 88 * MB);  // 16MB  => 104MB total
  unsigned short* qr16   = h16;     // safe alias: h consumed by q-GEMM
  unsigned short* kr16   = enc16;   // safe alias: enc consumed by kv-GEMM
  unsigned short* attn16 = qraw;    // safe alias: qraw consumed by rope-q

  // 1) rmsnorm(x) -> bf16, converts
  rmsnorm_to_bf16_k<<<WG_M, 256, 0, stream>>>(x, nw, h16, WG_D);
  conv_bf16_k<<<(WG_M * WG_D + 255) / 256, 256, 0, stream>>>(enc, enc16, WG_M * WG_D);
  conv_bf16_k<<<(WG_D * WG_D + 255) / 256, 256, 0, stream>>>(Wq,  wq16,  WG_D * WG_D);
  conv_bf16_k<<<(2 * WG_D * WG_D + 255) / 256, 256, 0, stream>>>(Wkv, wkv16, 2 * WG_D * WG_D);
  conv_bf16_k<<<(WG_D * WG_D + 255) / 256, 256, 0, stream>>>(Wo,  wo16,  WG_D * WG_D);

  // 2) projections: q = h @ Wq^T ; kv = enc @ Wkv^T
  {
    dim3 g(WG_D / 128, WG_M / 64);
    gemm_bf16_wmma_k<<<g, 128, 0, stream>>>(h16, wq16, WG_M, WG_D, WG_D,
                                            qraw, nullptr, nullptr);
  }
  {
    dim3 g(2 * WG_D / 128, WG_M / 64);
    gemm_bf16_wmma_k<<<g, 128, 0, stream>>>(enc16, wkv16, WG_M, 2 * WG_D, WG_D,
                                            kvraw, nullptr, nullptr);
  }

  // 3) RoPE + transpose to [B,H,T,D]
  const int ropeThreads = WG_B * WG_H * WG_T * 32;   // 4.19M
  rope_transpose_k<<<ropeThreads / 256, 256, 0, stream>>>(
      qraw, WG_D, 0, qr16, WG_T, /*doRope=*/1, doff, /*useOff=*/1);
  rope_transpose_k<<<ropeThreads / 256, 256, 0, stream>>>(
      kvraw, 2 * WG_D, 0, kr16, WG_T, /*doRope=*/1, doff, /*useOff=*/0);
  rope_transpose_k<<<ropeThreads / 256, 256, 0, stream>>>(
      kvraw, 2 * WG_D, WG_D, v16p, WG_T, /*doRope=*/0, doff, /*useOff=*/0);

  // 4) flash attention
  {
    dim3 g(WG_T / 64, WG_B * WG_H);
    flash_attn_k<<<g, 128, 0, stream>>>(qr16, kr16, v16p, attn16, WG_T, WG_T);
  }

  // 5) out = residual + attn @ Wo^T
  {
    dim3 g(WG_D / 128, WG_M / 64);
    gemm_bf16_wmma_k<<<g, 128, 0, stream>>>(attn16, wo16, WG_M, WG_D, WG_D,
                                            nullptr, out, x);
  }
}
